// Attention_39453569581282
// MI455X (gfx1250) — compile-verified, matrix operand order checked by reading, not run on hardware
//
#include <hip/hip_runtime.h>

typedef __attribute__((ext_vector_type(16))) __bf16 v16bf;
typedef __attribute__((ext_vector_type(8)))  __bf16 v8bf;
typedef __attribute__((ext_vector_type(8)))  float  v8f;

#define HEADS 8
#define DH 64
#define BATCH 2
#define SEQ 2048
#define DIM 512
#define INNER 512
#define QK_SCALE 0.125f   // 64^-0.5

static __device__ __forceinline__ v16bf cat8(v8bf a, v8bf b) {
    v16bf r;
#pragma unroll
    for (int i = 0; i < 8; i++) { r[i] = a[i]; r[i + 8] = b[i]; }
    return r;
}

// Fragment loader for v_wmma_f32_16x16x32_bf16 (A or B side).
// Memory layout: "row" (M for A, N for B) major with leading dim ld, contiguous K.
// Per ISA 7.12.2: lane holds row = lane&15; K chunks [kb..kb+7] and [16+kb..16+kb+7],
// kb = 8*(lane>>4). Both chunks are 16-byte aligned vector loads.
static __device__ __forceinline__ v16bf load_frag(const __bf16* base, int ld, int row,
                                                  int kb, int k0) {
    const __bf16* p = base + (size_t)row * ld + k0 + kb;
    v8bf lo = *(const v8bf*)(p);
    v8bf hi = *(const v8bf*)(p + 16);
    return cat8(lo, hi);
}

// ---------------- kernel 1: fp32 -> bf16 conversion (+ weight transposes) -------------
__global__ void convert_kernel(const float* __restrict__ x, const float* __restrict__ wqkv,
                               const float* __restrict__ wout,
                               __bf16* __restrict__ xb, __bf16* __restrict__ wqkv_t,
                               __bf16* __restrict__ wout_t) {
    int stride = gridDim.x * blockDim.x;
    int tid = blockIdx.x * blockDim.x + threadIdx.x;
    const int nx = BATCH * SEQ * DIM;
    for (int i = tid; i < nx; i += stride) xb[i] = (__bf16)x[i];
    const int nq = DIM * 3 * INNER;
    for (int i = tid; i < nq; i += stride) {
        int k = i / (3 * INNER), c = i % (3 * INNER);
        wqkv_t[(size_t)c * DIM + k] = (__bf16)wqkv[i];   // [col][k]
    }
    const int no = INNER * DIM;
    for (int i = tid; i < no; i += stride) {
        int k = i / DIM, c = i % DIM;
        wout_t[(size_t)c * INNER + k] = (__bf16)wout[i]; // [col][k]
    }
}

// ---------------- kernel 2: mask = softmax(adj, axis=-1), fp32 ----------------------
__global__ void mask_softmax_kernel(const float* __restrict__ adj, float* __restrict__ mask) {
    const int row = blockIdx.x;                    // 0 .. BATCH*SEQ-1
    const float* a = adj + (size_t)row * SEQ;
    float* o = mask + (size_t)row * SEQ;
    __shared__ float red[256];
    float mx = -1e30f;
    for (int j = threadIdx.x; j < SEQ; j += 256) mx = fmaxf(mx, a[j]);
    red[threadIdx.x] = mx; __syncthreads();
    for (int s = 128; s > 0; s >>= 1) {
        if (threadIdx.x < s) red[threadIdx.x] = fmaxf(red[threadIdx.x], red[threadIdx.x + s]);
        __syncthreads();
    }
    mx = red[0]; __syncthreads();
    float sm = 0.f;
    for (int j = threadIdx.x; j < SEQ; j += 256) sm += __expf(a[j] - mx);
    red[threadIdx.x] = sm; __syncthreads();
    for (int s = 128; s > 0; s >>= 1) {
        if (threadIdx.x < s) red[threadIdx.x] += red[threadIdx.x + s];
        __syncthreads();
    }
    float inv = 1.f / red[0];
    for (int j = threadIdx.x; j < SEQ; j += 256) o[j] = __expf(a[j] - mx) * inv;
}

// ---------------- kernel 3: qkv = x @ w_qkv (32x32 wave tile), scatter Q/K/V^T ------
__global__ __launch_bounds__(256) void qkv_gemm_kernel(const __bf16* __restrict__ xb,
                                                       const __bf16* __restrict__ wt,
                                                       __bf16* __restrict__ Qb,
                                                       __bf16* __restrict__ Kb,
                                                       __bf16* __restrict__ Vt) {
    const int wave = blockIdx.x * 8 + (threadIdx.x >> 5);
    const int lane = threadIdx.x & 31;
    const int row16 = lane & 15, lh = lane >> 4, kb = lh * 8;
    const int NT = (3 * INNER) / 32;               // 48 tiles of N (32 wide)
    const int mt = wave / NT, nt = wave % NT;
    const __bf16* Abase = xb + (size_t)mt * 32 * DIM;
    const __bf16* Bbase = wt + (size_t)nt * 32 * DIM;
    v8f acc[2][2];
    v8f zero = {};
#pragma unroll
    for (int i = 0; i < 2; i++)
#pragma unroll
        for (int j = 0; j < 2; j++) acc[i][j] = zero;
#pragma unroll 2
    for (int k0 = 0; k0 < DIM; k0 += 32) {
        v16bf a0 = load_frag(Abase, DIM, row16, kb, k0);
        v16bf a1 = load_frag(Abase, DIM, 16 + row16, kb, k0);
        v16bf b0 = load_frag(Bbase, DIM, row16, kb, k0);
        v16bf b1 = load_frag(Bbase, DIM, 16 + row16, kb, k0);
        acc[0][0] = __builtin_amdgcn_wmma_f32_16x16x32_bf16(false, a0, false, b0, (short)0,
                                                            acc[0][0], false, false);
        acc[0][1] = __builtin_amdgcn_wmma_f32_16x16x32_bf16(false, a0, false, b1, (short)0,
                                                            acc[0][1], false, false);
        acc[1][0] = __builtin_amdgcn_wmma_f32_16x16x32_bf16(false, a1, false, b0, (short)0,
                                                            acc[1][0], false, false);
        acc[1][1] = __builtin_amdgcn_wmma_f32_16x16x32_bf16(false, a1, false, b1, (short)0,
                                                            acc[1][1], false, false);
    }
#pragma unroll
    for (int ti = 0; ti < 2; ti++)
#pragma unroll
        for (int tj = 0; tj < 2; tj++)
#pragma unroll
            for (int r = 0; r < 8; r++) {
                int grow = mt * 32 + ti * 16 + r + 8 * lh;   // C: VGPR r -> M=r+8*(lane>>4)
                int gcol = nt * 32 + tj * 16 + row16;        //    lane&15 -> N
                int bb = grow >> 11, i = grow & (SEQ - 1);
                int region = gcol / INNER, w = gcol % INNER;
                int h = w / DH, d = w % DH;
                size_t bh = (size_t)bb * HEADS + h;
                float v = acc[ti][tj][r];
                if (region == 0)      Qb[(bh * SEQ + i) * DH + d] = (__bf16)(v * QK_SCALE);
                else if (region == 1) Kb[(bh * SEQ + i) * DH + d] = (__bf16)v;
                else                  Vt[(bh * DH + d) * SEQ + i] = (__bf16)v;  // transposed
            }
}

// ---------------- kernel 4: fused flash attention (one wave per 16 query rows) ------
__global__ __launch_bounds__(256) void attn_kernel(const __bf16* __restrict__ Qb,
                                                   const __bf16* __restrict__ Kb,
                                                   const __bf16* __restrict__ Vt,
                                                   const float* __restrict__ mask,
                                                   __bf16* __restrict__ aout) {
    __shared__ __attribute__((aligned(16))) __bf16 Pl[8][16 * 32];  // per-wave P tile
    const int wv = threadIdx.x >> 5;
    const int wave = blockIdx.x * 8 + wv;
    const int lane = threadIdx.x & 31;
    const int row16 = lane & 15, lh = lane >> 4, kb = lh * 8;
    const int mt = wave & 127;                     // 128 row-tiles per (b,h)
    const int bh = wave >> 7;                      // 0..15
    const int bb = bh >> 3;
    const __bf16* Qbase = Qb + ((size_t)bh * SEQ + mt * 16) * DH;
    const __bf16* Kbase = Kb + (size_t)bh * SEQ * DH;
    const __bf16* Vbase = Vt + (size_t)bh * DH * SEQ;
    const float*  mrow  = mask + ((size_t)bb * SEQ + mt * 16) * SEQ;
    __bf16* P = &Pl[wv][0];

    // Q fragments for both K-steps (d = 0..31, 32..63), loaded once. SCALE pre-folded.
    v16bf q0 = load_frag(Qbase, DH, row16, kb, 0);
    v16bf q1 = load_frag(Qbase, DH, row16, kb, 32);

    float mi[8], li[8];
    v8f acc[4];
    v8f zero = {};
#pragma unroll
    for (int r = 0; r < 8; r++) { mi[r] = -1e30f; li[r] = 0.f; }
#pragma unroll
    for (int f = 0; f < 4; f++) acc[f] = zero;

    for (int j0 = 0; j0 < SEQ; j0 += 32) {
        // prefetch next K/V tiles (global_prefetch_b8): one 128B line per lane covers
        // the 32x64 K tile; two strided lines per lane cover the 64-row V^T slab.
        if (j0 + 32 < SEQ) {
            __builtin_prefetch(Kbase + ((size_t)(j0 + 32)) * DH + lane * 64, 0, 0);
            __builtin_prefetch(Vbase + (size_t)(lane * 2) * SEQ + j0 + 32, 0, 0);
            __builtin_prefetch(Vbase + (size_t)(lane * 2 + 1) * SEQ + j0 + 32, 0, 0);
        }
        float p0[8], p1[8];
#pragma unroll
        for (int t = 0; t < 2; t++) {
            int j = j0 + t * 16;
            v8f s = zero;
            v16bf bk0 = load_frag(Kbase + (size_t)j * DH, DH, row16, kb, 0);
            s = __builtin_amdgcn_wmma_f32_16x16x32_bf16(false, q0, false, bk0, (short)0, s,
                                                        false, false);
            v16bf bk1 = load_frag(Kbase + (size_t)j * DH, DH, row16, kb, 32);
            s = __builtin_amdgcn_wmma_f32_16x16x32_bf16(false, q1, false, bk1, (short)0, s,
                                                        false, false);
            float* pd = t ? p1 : p0;
#pragma unroll
            for (int r = 0; r < 8; r++) {
                int M = r + 8 * lh;
                pd[r] = mrow[(size_t)M * SEQ + j + row16] * s[r];  // mask * (scaled dots)
            }
        }
        // online softmax over the 16x32 slab; row lives in a 16-lane half-group
        float corr[8];
#pragma unroll
        for (int r = 0; r < 8; r++) {
            float mx = fmaxf(p0[r], p1[r]);
#pragma unroll
            for (int sh = 1; sh < 16; sh <<= 1) mx = fmaxf(mx, __shfl_xor(mx, sh, 32));
            float nm = fmaxf(mi[r], mx);
            corr[r] = __expf(mi[r] - nm);
            float e0 = __expf(p0[r] - nm);
            float e1 = __expf(p1[r] - nm);
            float rs = e0 + e1;
#pragma unroll
            for (int sh = 1; sh < 16; sh <<= 1) rs += __shfl_xor(rs, sh, 32);
            li[r] = li[r] * corr[r] + rs;
            mi[r] = nm;
            p0[r] = e0; p1[r] = e1;
        }
#pragma unroll
        for (int f = 0; f < 4; f++)
#pragma unroll
            for (int r = 0; r < 8; r++) acc[f][r] *= corr[r];
        // stage exp(P) 16x32 bf16 through this wave's LDS slice (C-layout -> A-layout)
#pragma unroll
        for (int r = 0; r < 8; r++) {
            int M = r + 8 * lh;
            P[M * 32 + row16]      = (__bf16)p0[r];
            P[M * 32 + 16 + row16] = (__bf16)p1[r];
        }
        asm volatile("s_wait_dscnt 0" ::: "memory");  // same-wave LDS store->load order
        v16bf pf = load_frag(P, 32, row16, kb, 0);
#pragma unroll
        for (int f = 0; f < 4; f++) {
            v16bf vf = load_frag(Vbase + j0, SEQ, f * 16 + row16, kb, 0);
            acc[f] = __builtin_amdgcn_wmma_f32_16x16x32_bf16(false, pf, false, vf, (short)0,
                                                             acc[f], false, false);
        }
    }
    // normalize and emit bf16 [b][i][h*64+d] for the output projection
    __bf16* orow = aout + ((size_t)bb * SEQ + mt * 16) * INNER + (bh & 7) * DH;
#pragma unroll
    for (int f = 0; f < 4; f++)
#pragma unroll
        for (int r = 0; r < 8; r++) {
            int M = r + 8 * lh;
            orow[(size_t)M * INNER + f * 16 + row16] = (__bf16)(acc[f][r] / li[r]);
        }
}

// ---------------- kernel 5: out = attn_out @ w_out + b_out (32x32 wave tile) --------
__global__ __launch_bounds__(256) void out_gemm_kernel(const __bf16* __restrict__ aout,
                                                       const __bf16* __restrict__ wt,
                                                       const float* __restrict__ bias,
                                                       float* __restrict__ out) {
    const int wave = blockIdx.x * 8 + (threadIdx.x >> 5);
    const int lane = threadIdx.x & 31;
    const int row16 = lane & 15, lh = lane >> 4, kb = lh * 8;
    const int NT = DIM / 32;                       // 16 tiles of N (32 wide)
    const int mt = wave / NT, nt = wave % NT;
    const __bf16* Abase = aout + (size_t)mt * 32 * INNER;
    const __bf16* Bbase = wt + (size_t)nt * 32 * INNER;
    v8f acc[2][2];
    v8f zero = {};
#pragma unroll
    for (int i = 0; i < 2; i++)
#pragma unroll
        for (int j = 0; j < 2; j++) acc[i][j] = zero;
#pragma unroll 2
    for (int k0 = 0; k0 < INNER; k0 += 32) {
        v16bf a0 = load_frag(Abase, INNER, row16, kb, k0);
        v16bf a1 = load_frag(Abase, INNER, 16 + row16, kb, k0);
        v16bf b0 = load_frag(Bbase, INNER, row16, kb, k0);
        v16bf b1 = load_frag(Bbase, INNER, 16 + row16, kb, k0);
        acc[0][0] = __builtin_amdgcn_wmma_f32_16x16x32_bf16(false, a0, false, b0, (short)0,
                                                            acc[0][0], false, false);
        acc[0][1] = __builtin_amdgcn_wmma_f32_16x16x32_bf16(false, a0, false, b1, (short)0,
                                                            acc[0][1], false, false);
        acc[1][0] = __builtin_amdgcn_wmma_f32_16x16x32_bf16(false, a1, false, b0, (short)0,
                                                            acc[1][0], false, false);
        acc[1][1] = __builtin_amdgcn_wmma_f32_16x16x32_bf16(false, a1, false, b1, (short)0,
                                                            acc[1][1], false, false);
    }
#pragma unroll
    for (int ti = 0; ti < 2; ti++)
#pragma unroll
        for (int tj = 0; tj < 2; tj++)
#pragma unroll
            for (int r = 0; r < 8; r++) {
                int grow = mt * 32 + ti * 16 + r + 8 * lh;
                int gcol = nt * 32 + tj * 16 + row16;
                out[(size_t)grow * DIM + gcol] = acc[ti][tj][r] + bias[gcol];
            }
}

extern "C" void kernel_launch(void* const* d_in, const int* in_sizes, int n_in,
                              void* d_out, int out_size, void* d_ws, size_t ws_size,
                              hipStream_t stream) {
    const float* x    = (const float*)d_in[0];
    const float* adj  = (const float*)d_in[1];
    const float* wqkv = (const float*)d_in[2];
    const float* wout = (const float*)d_in[3];
    const float* bout = (const float*)d_in[4];
    float* out = (float*)d_out;

    char* ws = (char*)d_ws;
    size_t off = 0;
    auto alloc = [&](size_t bytes) -> void* {
        void* p = ws + off;
        off += (bytes + 255) & ~(size_t)255;
        return p;
    };
    __bf16* xb     = (__bf16*)alloc((size_t)BATCH * SEQ * DIM * 2);
    __bf16* wqkv_t = (__bf16*)alloc((size_t)3 * INNER * DIM * 2);
    __bf16* wout_t = (__bf16*)alloc((size_t)DIM * INNER * 2);
    __bf16* Qb     = (__bf16*)alloc((size_t)BATCH * HEADS * SEQ * DH * 2);
    __bf16* Kb     = (__bf16*)alloc((size_t)BATCH * HEADS * SEQ * DH * 2);
    __bf16* Vt     = (__bf16*)alloc((size_t)BATCH * HEADS * SEQ * DH * 2);
    float*  mask   = (float*)alloc((size_t)BATCH * SEQ * SEQ * 4);
    __bf16* aout   = (__bf16*)alloc((size_t)BATCH * SEQ * INNER * 2);

    convert_kernel<<<1024, 256, 0, stream>>>(x, wqkv, wout, xb, wqkv_t, wout_t);
    mask_softmax_kernel<<<BATCH * SEQ, 256, 0, stream>>>(adj, mask);
    // (BATCH*SEQ/32) * (3*INNER/32) = 128*48 wave-tiles, 8 waves/block
    qkv_gemm_kernel<<<(128 * 48) / 8, 256, 0, stream>>>(xb, wqkv_t, Qb, Kb, Vt);
    // BATCH*HEADS*(SEQ/16) = 2048 wave-tiles
    attn_kernel<<<2048 / 8, 256, 0, stream>>>(Qb, Kb, Vt, mask, aout);
    // (BATCH*SEQ/32) * (DIM/32) = 128*16 wave-tiles
    out_gemm_kernel<<<(128 * 16) / 8, 256, 0, stream>>>(aout, wout_t, bout, out);
}
